// ImaginationCore_45303315038686
// MI455X (gfx1250) — matrix operand order repaired and checked by reading, not compiled, not run on hardware
//
#include <hip/hip_runtime.h>
#include <hip/hip_bf16.h>
#include <math.h>

// ---------------- problem dimensions ----------------
#define B_DIM   2048
#define S_DIM   12288
#define H_DIM   1024
#define NA_DIM  2312
#define NAP_DIM 2368          // NA padded to multiple of 64 for 64x64 wave tiles
#define T_STEPS 17

// state layout indices
#define IDX_MY_DMG 110
#define IDX_MY_ACC 111
#define IDX_MY_VLN 112
#define IDX_EN_DMG 160
#define IDX_EN_ACC 161
#define IDX_EN_VLN 162

typedef __attribute__((ext_vector_type(16))) __bf16 bf16x16;
typedef __attribute__((ext_vector_type(8)))  __bf16 bf16x8;
typedef __attribute__((ext_vector_type(8)))  float  f32x8;

// ---------------------------------------------------------------------------
// helpers
// ---------------------------------------------------------------------------
__device__ __forceinline__ bf16x16 make_bf16x16(bf16x8 lo, bf16x8 hi) {
  bf16x16 r;
#pragma unroll
  for (int i = 0; i < 8; ++i) { r[i] = lo[i]; r[i + 8] = hi[i]; }
  return r;
}

__device__ __forceinline__ float sigmoidf(float x) {
  return 1.0f / (1.0f + __expf(-x));
}

__device__ __forceinline__ int argmax3(const float* p) {
  int idx = 0; float v = p[0];
  if (p[1] > v) { v = p[1]; idx = 1; }
  if (p[2] > v) { v = p[2]; idx = 2; }
  return idx;
}

// ---------------------------------------------------------------------------
// WMMA core: one wave (32 threads) computes a 64(M) x 64(N) output tile
// as a 4x4 grid of V_WMMA_F32_16X16X32_BF16 accumulators.
// A  : row-major [M, K] bf16       (lda == K)
// BT : row-major [N, K] bf16       (i.e. B transposed; ldb == K)
// Fragment layouts used here (contiguous k order, same for A and B; not the ISA table order):
//   A 16x32 bf16 : lane m=l&15; lo-lanes K=[k..k+15], hi-lanes K=[k+16..k+31]
//   B 32x16 bf16 : lane n=l&15; lo-lanes K=[k..k+15], hi-lanes K=[k+16..k+31]
// ---------------------------------------------------------------------------
__device__ __forceinline__ void wave_gemm_64x64(const __bf16* __restrict__ A,
                                                const __bf16* __restrict__ BT,
                                                int K, int m0, int n0,
                                                f32x8 acc[4][4]) {
  const int lane = threadIdx.x & 31;
  const int sub  = lane & 15;
  const int alo  = (lane >> 4) << 3;    // 0 or 8
  const int blo  = (lane >> 4) << 4;    // 0 or 16

  const __bf16* aptr[4];
  const __bf16* bptr[4];
#pragma unroll
  for (int mi = 0; mi < 4; ++mi)
    aptr[mi] = A + (long)(m0 + mi * 16 + sub) * K;
#pragma unroll
  for (int ni = 0; ni < 4; ++ni)
    bptr[ni] = BT + (long)(n0 + ni * 16 + sub) * K;

  for (int k = 0; k < K; k += 32) {
    // Speculative prefetch of upcoming K-slices (OOB prefetch is dropped).
    __builtin_prefetch(aptr[0] + k + 512, 0, 1);
    __builtin_prefetch(bptr[0] + k + 512, 0, 1);

    bf16x16 afr[4];
#pragma unroll
    for (int mi = 0; mi < 4; ++mi) {
      afr[mi] = *(const bf16x16*)(aptr[mi] + k + blo);
    }
#pragma unroll
    for (int ni = 0; ni < 4; ++ni) {
      const bf16x16 bfr = *(const bf16x16*)(bptr[ni] + k + blo);
#pragma unroll
      for (int mi = 0; mi < 4; ++mi)
        acc[mi][ni] = __builtin_amdgcn_wmma_f32_16x16x32_bf16(
            false, afr[mi], false, bfr, (short)0, acc[mi][ni],
            false, false);
    }
  }
}

// ---------------------------------------------------------------------------
// GEMM1: h = tanh(state_bf @ W1 + act_emb[action] + b1)  -> h_bf [B,H]
// ---------------------------------------------------------------------------
__global__ void __launch_bounds__(32)
gemm1_kernel(const __bf16* __restrict__ Sb, const __bf16* __restrict__ W1T,
             const float* __restrict__ b1, const float* __restrict__ act_emb,
             const int* __restrict__ action, __bf16* __restrict__ Hb) {
  f32x8 acc[4][4] = {};
  const int m0 = blockIdx.x * 64, n0 = blockIdx.y * 64;
  wave_gemm_64x64(Sb, W1T, S_DIM, m0, n0, acc);

  const int lane = threadIdx.x & 31;
  const int hi = lane >> 4, nc = lane & 15;
#pragma unroll
  for (int mi = 0; mi < 4; ++mi)
#pragma unroll
    for (int ni = 0; ni < 4; ++ni) {
      const int n = n0 + ni * 16 + nc;
      const float bb = b1[n];
#pragma unroll
      for (int r = 0; r < 8; ++r) {
        const int m = m0 + mi * 16 + (hi << 3) + r;
        const float v = acc[mi][ni][r] + bb +
                        act_emb[(long)action[m] * H_DIM + n];
        Hb[(long)m * H_DIM + n] = (__bf16)tanhf(v);
      }
    }
}

// ---------------------------------------------------------------------------
// GEMM2: new_state = sigmoid(h_bf @ W2); masked write to f32 + bf16 state
// ---------------------------------------------------------------------------
__global__ void __launch_bounds__(32)
gemm2_kernel(const __bf16* __restrict__ Hb, const __bf16* __restrict__ W2T,
             const unsigned char* __restrict__ active,
             float* __restrict__ Sf, __bf16* __restrict__ Sb) {
  f32x8 acc[4][4] = {};
  const int m0 = blockIdx.x * 64, n0 = blockIdx.y * 64;
  wave_gemm_64x64(Hb, W2T, H_DIM, m0, n0, acc);

  const int lane = threadIdx.x & 31;
  const int hi = lane >> 4, nc = lane & 15;
#pragma unroll
  for (int mi = 0; mi < 4; ++mi)
#pragma unroll
    for (int ni = 0; ni < 4; ++ni) {
      const int n = n0 + ni * 16 + nc;
#pragma unroll
      for (int r = 0; r < 8; ++r) {
        const int m = m0 + mi * 16 + (hi << 3) + r;
        if (active[m]) {
          const float v = sigmoidf(acc[mi][ni][r]);
          Sf[(long)m * S_DIM + n] = v;
          Sb[(long)m * S_DIM + n] = (__bf16)v;
        }
      }
    }
}

// ---------------------------------------------------------------------------
// GEMM-A: logits = state_bf @ Wa  -> f32 [B, NAP]
// ---------------------------------------------------------------------------
__global__ void __launch_bounds__(32)
gemma_kernel(const __bf16* __restrict__ Sb, const __bf16* __restrict__ WaT,
             float* __restrict__ logits) {
  f32x8 acc[4][4] = {};
  const int m0 = blockIdx.x * 64, n0 = blockIdx.y * 64;
  wave_gemm_64x64(Sb, WaT, S_DIM, m0, n0, acc);

  const int lane = threadIdx.x & 31;
  const int hi = lane >> 4, nc = lane & 15;
#pragma unroll
  for (int mi = 0; mi < 4; ++mi)
#pragma unroll
    for (int ni = 0; ni < 4; ++ni) {
      const int n = n0 + ni * 16 + nc;
#pragma unroll
      for (int r = 0; r < 8; ++r) {
        const int m = m0 + mi * 16 + (hi << 3) + r;
        logits[(long)m * NAP_DIM + n] = acc[mi][ni][r];
      }
    }
}

// ---------------------------------------------------------------------------
// small per-row kernels
// ---------------------------------------------------------------------------
__global__ void convert_transpose_kernel(const float* __restrict__ src,
                                         __bf16* __restrict__ dst,
                                         int R /*src rows=K*/, int C /*src cols*/,
                                         long total /* = Cpad * R */) {
  long i = (long)blockIdx.x * blockDim.x + threadIdx.x;
  if (i >= total) return;
  const int k = (int)(i % R);
  const int n = (int)(i / R);
  dst[i] = (n < C) ? (__bf16)src[(long)k * C + n] : (__bf16)0.0f;
}

__global__ void init_state_kernel(const float* __restrict__ in,
                                  float* __restrict__ Sf,
                                  __bf16* __restrict__ Sb, long n) {
  long i = (long)blockIdx.x * blockDim.x + threadIdx.x;
  if (i >= n) return;
  const float v = in[i];
  Sf[i] = v;
  Sb[i] = (__bf16)v;
}

__global__ void init_meta_kernel(const float* __restrict__ init_state,
                                 const int* __restrict__ init_action,
                                 int* __restrict__ init_player,
                                 unsigned char* __restrict__ init_done,
                                 float* __restrict__ reward,
                                 int* __restrict__ action) {
  const int b = blockIdx.x * blockDim.x + threadIdx.x;
  if (b >= B_DIM) return;
  const float* row = init_state + (long)b * S_DIM;
  const int p = argmax3(row);
  const int w = argmax3(row + 3);
  init_player[b] = p;
  init_done[b] = ((p == 0) || (w > 0)) ? 1 : 0;
  reward[b] = 0.0f;
  action[b] = init_action[b];
}

__global__ void control_kernel(const float* __restrict__ Sf,
                               const int* __restrict__ init_player,
                               unsigned char* __restrict__ active, int step) {
  const int b = blockIdx.x * blockDim.x + threadIdx.x;
  if (b >= B_DIM) return;
  const float* row = Sf + (long)b * S_DIM;
  const int p = argmax3(row);
  const int w = argmax3(row + 3);
  const bool done = (p == 0) || (w > 0);
  const bool act = (step == 0) ? !done : (!done && (p != init_player[b]));
  active[b] = act ? 1 : 0;
}

__global__ void __launch_bounds__(256)
argmax_kernel(const float* __restrict__ logits, int* __restrict__ action) {
  const int row = blockIdx.x * 8 + (threadIdx.x >> 5);
  const int lane = threadIdx.x & 31;
  const float* L = logits + (long)row * NAP_DIM;
  float best = -INFINITY;
  int bi = 0;
  for (int c = lane; c < NA_DIM; c += 32) {
    const float v = L[c];
    if (v > best) { best = v; bi = c; }
  }
#pragma unroll
  for (int off = 16; off > 0; off >>= 1) {
    const float ov = __shfl_xor(best, off, 32);
    const int oi = __shfl_xor(bi, off, 32);
    if (ov > best || (ov == best && oi < bi)) { best = ov; bi = oi; }
  }
  if (lane == 0) action[row] = bi;
}

__global__ void reward_step_kernel(const float* __restrict__ Sf,
                                   const unsigned char* __restrict__ active,
                                   float* __restrict__ reward) {
  const int b = blockIdx.x * blockDim.x + threadIdx.x;
  if (b >= B_DIM) return;
  if (!active[b]) return;
  const float* s = Sf + (long)b * S_DIM;
  reward[b] += 1000.0f * ((s[IDX_EN_VLN] - s[IDX_MY_VLN]) +
                          (s[IDX_EN_DMG] - s[IDX_MY_DMG]));
}

__global__ void final_kernel(const float* __restrict__ Sf,
                             const float* __restrict__ reward,
                             const unsigned char* __restrict__ init_done,
                             float* __restrict__ out_reward,
                             float* __restrict__ out_done) {
  const int b = blockIdx.x * blockDim.x + threadIdx.x;
  if (b >= B_DIM) return;
  const float* s = Sf + (long)b * S_DIM;
  const int p = argmax3(s);
  const int w = argmax3(s + 3);
  const bool done = (p == 0) || (w > 0);
  float r = reward[b];
  const bool idone = init_done[b] != 0;
  if (!idone) r += -0.01f;
  if (done && !idone) {
    r += 1000.0f * ((s[IDX_EN_VLN] - s[IDX_MY_VLN]) +
                    (s[IDX_EN_DMG] - s[IDX_MY_DMG])) +
         1000.0f * (s[IDX_EN_ACC] - s[IDX_MY_ACC]);
  }
  out_reward[b] = r;
  out_done[b] = done ? 1.0f : 0.0f;
}

// ---------------------------------------------------------------------------
// host driver
// ---------------------------------------------------------------------------
extern "C" void kernel_launch(void* const* d_in, const int* in_sizes, int n_in,
                              void* d_out, int out_size, void* d_ws,
                              size_t ws_size, hipStream_t stream) {
  (void)in_sizes; (void)n_in; (void)out_size; (void)ws_size;

  const float* in_state  = (const float*)d_in[0];
  const int*   in_action = (const int*)d_in[1];
  const float* W1        = (const float*)d_in[2];
  const float* b1        = (const float*)d_in[3];
  const float* W2        = (const float*)d_in[4];
  const float* act_emb   = (const float*)d_in[5];
  const float* Wa        = (const float*)d_in[6];

  // ---- workspace carve ----
  char* ws = (char*)d_ws;
  size_t off = 0;
  auto carve = [&](size_t bytes) -> void* {
    void* p = ws + off;
    off = (off + bytes + 255) & ~(size_t)255;
    return p;
  };
  float*         Sf      = (float*)        carve((size_t)B_DIM * S_DIM * 4);
  __bf16*        Sb      = (__bf16*)       carve((size_t)B_DIM * S_DIM * 2);
  __bf16*        Hb      = (__bf16*)       carve((size_t)B_DIM * H_DIM * 2);
  __bf16*        W1T     = (__bf16*)       carve((size_t)H_DIM * S_DIM * 2);
  __bf16*        W2T     = (__bf16*)       carve((size_t)S_DIM * H_DIM * 2);
  __bf16*        WaT     = (__bf16*)       carve((size_t)NAP_DIM * S_DIM * 2);
  float*         logits  = (float*)        carve((size_t)B_DIM * NAP_DIM * 4);
  int*           action  = (int*)          carve((size_t)B_DIM * 4);
  int*           iplayer = (int*)          carve((size_t)B_DIM * 4);
  unsigned char* idone   = (unsigned char*)carve((size_t)B_DIM);
  unsigned char* active  = (unsigned char*)carve((size_t)B_DIM);
  float*         reward  = (float*)        carve((size_t)B_DIM * 4);

  float* out_state  = (float*)d_out;
  float* out_reward = out_state + (size_t)B_DIM * S_DIM;
  float* out_done   = out_reward + B_DIM;

  // ---- one-time (per launch) bf16 transposed weights ----
  {
    long t1 = (long)H_DIM * S_DIM;
    convert_transpose_kernel<<<(unsigned)((t1 + 255) / 256), 256, 0, stream>>>(
        W1, W1T, S_DIM, H_DIM, t1);
    long t2 = (long)S_DIM * H_DIM;
    convert_transpose_kernel<<<(unsigned)((t2 + 255) / 256), 256, 0, stream>>>(
        W2, W2T, H_DIM, S_DIM, t2);
    long ta = (long)NAP_DIM * S_DIM;
    convert_transpose_kernel<<<(unsigned)((ta + 255) / 256), 256, 0, stream>>>(
        Wa, WaT, S_DIM, NA_DIM, ta);
    long ts = (long)B_DIM * S_DIM;
    init_state_kernel<<<(unsigned)((ts + 255) / 256), 256, 0, stream>>>(
        in_state, Sf, Sb, ts);
    init_meta_kernel<<<B_DIM / 256, 256, 0, stream>>>(
        in_state, in_action, iplayer, idone, reward, action);
  }

  const dim3 g1(B_DIM / 64, H_DIM / 64);     // (32, 16)
  const dim3 g2(B_DIM / 64, S_DIM / 64);     // (32, 192)
  const dim3 ga(B_DIM / 64, NAP_DIM / 64);   // (32, 37)

  // ---- rollout ----
  for (int i = 0; i < T_STEPS; ++i) {
    if (i > 0) {
      gemma_kernel<<<ga, 32, 0, stream>>>(Sb, WaT, logits);
      argmax_kernel<<<B_DIM / 8, 256, 0, stream>>>(logits, action);
    }
    control_kernel<<<B_DIM / 256, 256, 0, stream>>>(Sf, iplayer, active, i);
    gemm1_kernel<<<g1, 32, 0, stream>>>(Sb, W1T, b1, act_emb, action, Hb);
    gemm2_kernel<<<g2, 32, 0, stream>>>(Hb, W2T, active, Sf, Sb);
    reward_step_kernel<<<B_DIM / 256, 256, 0, stream>>>(Sf, active, reward);
  }

  // ---- finalization + outputs ----
  final_kernel<<<B_DIM / 256, 256, 0, stream>>>(Sf, reward, idone, out_reward,
                                                out_done);
  hipMemcpyAsync(out_state, Sf, (size_t)B_DIM * S_DIM * 4,
                 hipMemcpyDeviceToDevice, stream);
}
